// SpatioTemporalModel_8632884265282
// MI455X (gfx1250) — matrix operand order repaired and checked by reading, not compiled, not run on hardware
//
#include <hip/hip_runtime.h>
#include <hip/hip_bf16.h>
#include <math.h>

// ---------------------------------------------------------------------------
// ChebGRU (K=3) spatio-temporal model for MI455X / gfx1250 (wave32, WMMA).
//
// Dense path: fused 6-pair GEMM per gate using v_wmma_f32_16x16x32_f16.
//   - Weights pre-packed into B-fragment layout (one 32B load/lane/fragment).
//   - Shared 16x128 A tiles staged in LDS once per block (24KB of the
//     320KB/WGP), fragments then fed from ds_load_b128 (8x less global
//     traffic than per-wave global loads).
// Sparse path: edge-parallel gather + global_atomic_add_f32 scatter
//   (problem is L2-resident on the 192MB L2: features+edges ~100MB).
// ---------------------------------------------------------------------------

#define N_NODES 10000
#define N_EDGES 320000
#define T_STEPS 10
#define HID 128
#define OUTC 64

typedef __attribute__((ext_vector_type(16))) _Float16 v16h;
typedef __attribute__((ext_vector_type(8)))  float    v8f;

// ---------------- sparse / elementwise kernels ----------------

__global__ void deg_kernel(const long long* __restrict__ src,
                           const long long* __restrict__ dst,
                           float* __restrict__ deg) {
  int e = blockIdx.x * 256 + threadIdx.x;
  if (e >= N_EDGES) return;
  long long s = src[e], d = dst[e];
  if (s != d) atomicAdd(deg + (int)s, 1.0f);
}

__global__ void edgew_kernel(const long long* __restrict__ src,
                             const long long* __restrict__ dst,
                             const float* __restrict__ deg,
                             float* __restrict__ w) {
  int e = blockIdx.x * 256 + threadIdx.x;
  if (e >= N_EDGES) return;
  long long s = src[e], d = dst[e];
  float ws = 0.0f;
  if (s != d) {
    float ds_ = deg[(int)s], dd_ = deg[(int)d];
    float is_ = (ds_ > 0.0f) ? rsqrtf(fmaxf(ds_, 1.0f)) : 0.0f;
    float id_ = (dd_ > 0.0f) ? rsqrtf(fmaxf(dd_, 1.0f)) : 0.0f;
    ws = -is_ * id_;
  }
  w[e] = ws;
}

// One wave (32 lanes) per edge; each lane handles 4 channels via float4.
__global__ __launch_bounds__(256) void agg_kernel(
    const float* __restrict__ w,
    const long long* __restrict__ src,
    const long long* __restrict__ dst,
    const float* __restrict__ X,
    float* __restrict__ Y) {
  int e    = blockIdx.x * 8 + (threadIdx.x >> 5);
  int lane = threadIdx.x & 31;
  if (e >= N_EDGES) return;
  float we = w[e];
  if (we == 0.0f) return;
  int s = (int)src[e], d = (int)dst[e];
  const float4* xs = (const float4*)(X + (size_t)s * HID);
  float*        yd = Y + (size_t)d * HID + lane * 4;
  float4 v = xs[lane];
  atomicAdd(yd + 0, we * v.x);
  atomicAdd(yd + 1, we * v.y);
  atomicAdd(yd + 2, we * v.z);
  atomicAdd(yd + 3, we * v.w);
}

__global__ void cvt16_kernel(const float* __restrict__ x,
                             _Float16* __restrict__ y, int n) {
  int i = blockIdx.x * 256 + threadIdx.x;
  if (i < n) y[i] = (_Float16)x[i];
}

// T2 = 2 * agg2 - base, stored as f16 for the WMMA GEMM.
__global__ void cheb2_kernel(const float* __restrict__ agg2,
                             const float* __restrict__ base,
                             _Float16* __restrict__ y, int n) {
  int i = blockIdx.x * 256 + threadIdx.x;
  if (i < n) y[i] = (_Float16)(2.0f * agg2[i] - base[i]);
}

// ---------------- weight packing into WMMA B-fragment layout ----------------
// Packed layout: out[((kc*ctiles + ct)*32 + lane)*16 + i]
//   = W[(kc*32 + (lane>>4)*8 + (i&7) + (i>>3)*16) * (ctiles*16) + ct*16 + (lane&15)]
// so each lane's 16-half fragment is one contiguous 32B load.
__global__ void pack_weight_kernel(const float* __restrict__ W,
                                   _Float16* __restrict__ out, int ctiles) {
  int idx = blockIdx.x * 256 + threadIdx.x;
  int total = 4 * ctiles * 512;
  if (idx >= total) return;
  int i    = idx & 15;
  int lane = (idx >> 4) & 31;
  int ct   = (idx >> 9) % ctiles;
  int kc   = idx / (512 * ctiles);
  int l  = lane & 15;
  int hi = lane >> 4;
  int krow = kc * 32 + hi * 8 + (i & 7) + ((i >> 3) << 4);
  int col  = ct * 16 + l;
  out[idx] = (_Float16)W[(size_t)krow * (ctiles * 16) + col];
}

// ---------------- fused 6-pair WMMA GEMM with GRU epilogues ----------------
// Each block: 16 rows x 128 cols; 8 waves, each wave one 16x16 tile.
// The six 16x128 f16 A tiles are staged once in LDS (24KB), then each wave
// accumulates sum_{p=0..5} A_p @ W_p (K=128 each -> 24 WMMAs / tile) with
// A fragments from ds_load_b128 and pre-packed B fragments from global.
// mode 0: out = sigmoid(pre + b0 + b1)                          (z gate)
// mode 1: out = aux1 * sigmoid(pre + b0 + b1)                   (hr = h*r)
// mode 2: out = aux1*aux2 + (1-aux1)*tanh(pre + b0 + b1)        (h update)
__global__ __launch_bounds__(256) void gemm_fused6_kernel(
    const _Float16* __restrict__ A0, const _Float16* __restrict__ A1,
    const _Float16* __restrict__ A2, const _Float16* __restrict__ A3,
    const _Float16* __restrict__ A4, const _Float16* __restrict__ A5,
    const _Float16* __restrict__ P0, const _Float16* __restrict__ P1,
    const _Float16* __restrict__ P2, const _Float16* __restrict__ P3,
    const _Float16* __restrict__ P4, const _Float16* __restrict__ P5,
    const float* __restrict__ b0, const float* __restrict__ b1,
    const float* __restrict__ aux1, const float* __restrict__ aux2,
    float* __restrict__ out, int mode) {
  __shared__ _Float16 ldsA[6][16 * HID];   // 6 x 4KB = 24KB per block

  const int tid  = threadIdx.x;
  const int wave = tid >> 5;
  const int lane = tid & 31;
  const int l    = lane & 15;
  const int hi   = lane >> 4;
  const int row0 = blockIdx.x * 16;
  const int ct   = wave;           // 8 column tiles of 16 -> 128 cols
  const int col0 = ct * 16;

  const _Float16* As[6] = {A0, A1, A2, A3, A4, A5};
  const _Float16* Ps[6] = {P0, P1, P2, P3, P4, P5};

  // Cooperative stage: 256 threads x 16B per pair covers 16x128 halves.
  const int srow = tid >> 4;       // 0..15
  const int sseg = tid & 15;       // 0..15 (8 halves each)
#pragma unroll
  for (int p = 0; p < 6; ++p) {
    const float4* gsrc =
        (const float4*)(As[p] + (size_t)(row0 + srow) * HID + sseg * 8);
    *(float4*)&ldsA[p][srow * HID + sseg * 8] = *gsrc;
  }
  __syncthreads();

  v8f acc = {};
#pragma unroll
  for (int p = 0; p < 6; ++p) {
    const _Float16* P = Ps[p];
#pragma unroll
    for (int kc = 0; kc < 4; ++kc) {
      const int kk = kc * 32;
      // A fragment (16x32 f16) from LDS: lane l/hi layout per ISA 7.12.2
      const _Float16* arow = &ldsA[p][l * HID + kk + hi * 8];
      v16h a, b;
#pragma unroll
      for (int i = 0; i < 8; ++i) {
        a[i]     = arow[i];        // K = kk + hi*8 + i
        a[8 + i] = arow[16 + i];   // K = kk + 16 + hi*8 + i
      }
      // B fragment: pre-packed, contiguous 32B per lane from global (L2-hot)
      const _Float16* bptr = P + ((size_t)((kc * 8 + ct) * 32 + lane)) * 16;
#pragma unroll
      for (int i = 0; i < 16; ++i) b[i] = bptr[i];
      acc = __builtin_amdgcn_wmma_f32_16x16x32_f16(
          false, a, false, b, (short)0, acc, false, false);
    }
  }

  // Epilogue: C layout — lane l, half hi: VGPR v -> (row0 + hi*8 + v, col0 + l)
#pragma unroll
  for (int v = 0; v < 8; ++v) {
    int row = row0 + hi * 8 + v;
    int col = col0 + l;
    size_t idx = (size_t)row * HID + col;
    float pre = acc[v] + b0[col] + b1[col];
    if (mode == 0) {
      out[idx] = 1.0f / (1.0f + __expf(-pre));
    } else if (mode == 1) {
      float r = 1.0f / (1.0f + __expf(-pre));
      out[idx] = aux1[idx] * r;                       // hr = h * r
    } else {
      float ht = tanhf(pre);
      float z  = aux1[idx];
      out[idx] = z * aux2[idx] + (1.0f - z) * ht;     // new h (in place)
    }
  }
}

// ---------------- output projection: out = h @ Wp + bp (N x 64) ----------------
__global__ __launch_bounds__(128) void gemm_proj_kernel(
    const _Float16* __restrict__ A, const _Float16* __restrict__ P,
    const float* __restrict__ bias, float* __restrict__ out) {
  __shared__ _Float16 ldsA[16 * HID];      // 4KB

  const int tid  = threadIdx.x;
  const int wave = tid >> 5;
  const int lane = tid & 31;
  const int l    = lane & 15;
  const int hi   = lane >> 4;
  const int row0 = blockIdx.x * 16;
  const int ct   = wave;           // 4 column tiles -> 64 cols
  const int col0 = ct * 16;

  // Stage 16x128 halves with 128 threads x 16 halves (two float4 each).
  {
    const float4* gsrc = (const float4*)(A + (size_t)row0 * HID) + tid * 2;
    float4* ldst = (float4*)ldsA + tid * 2;
    ldst[0] = gsrc[0];
    ldst[1] = gsrc[1];
  }
  __syncthreads();

  v8f acc = {};
#pragma unroll
  for (int kc = 0; kc < 4; ++kc) {
    const int kk = kc * 32;
    const _Float16* arow = &ldsA[l * HID + kk + hi * 8];
    v16h a, b;
#pragma unroll
    for (int i = 0; i < 8; ++i) {
      a[i]     = arow[i];
      a[8 + i] = arow[16 + i];
    }
    const _Float16* bptr = P + ((size_t)((kc * 4 + ct) * 32 + lane)) * 16;
#pragma unroll
    for (int i = 0; i < 16; ++i) b[i] = bptr[i];
    acc = __builtin_amdgcn_wmma_f32_16x16x32_f16(
        false, a, false, b, (short)0, acc, false, false);
  }
#pragma unroll
  for (int v = 0; v < 8; ++v) {
    int row = row0 + hi * 8 + v;
    int col = col0 + l;
    out[(size_t)row * OUTC + col] = acc[v] + bias[col];
  }
}

// ---------------- host-side orchestration ----------------

extern "C" void kernel_launch(void* const* d_in, const int* in_sizes, int n_in,
                              void* d_out, int out_size, void* d_ws, size_t ws_size,
                              hipStream_t stream) {
  (void)in_sizes; (void)n_in; (void)out_size; (void)ws_size;

  const float*     feats = (const float*)d_in[0];
  const long long* edges = (const long long*)d_in[1];
  const float* Wsrc[6] = {(const float*)d_in[2],  (const float*)d_in[4],
                          (const float*)d_in[6],  (const float*)d_in[8],
                          (const float*)d_in[10], (const float*)d_in[12]};
  const float* Bsrc[6] = {(const float*)d_in[3],  (const float*)d_in[5],
                          (const float*)d_in[7],  (const float*)d_in[9],
                          (const float*)d_in[11], (const float*)d_in[13]};
  const float* Wp = (const float*)d_in[14];
  const float* bp = (const float*)d_in[15];
  float* out = (float*)d_out;

  // -------- carve workspace --------
  char* wp_ = (char*)d_ws;
  auto carve = [&](size_t bytes) -> void* {
    void* r = (void*)wp_;
    wp_ += (bytes + 255) & ~(size_t)255;
    return r;
  };
  const size_t NC  = (size_t)N_NODES * HID;       // 1.28M elems
  const size_t NC4 = NC * sizeof(float);
  const size_t NC2 = NC * sizeof(_Float16);

  float* deg = (float*)carve(N_NODES * sizeof(float));
  float* ew  = (float*)carve(N_EDGES * sizeof(float));
  float* h   = (float*)carve(NC4);
  float* z   = (float*)carve(NC4);
  float* hr  = (float*)carve(NC4);
  float* t1f = (float*)carve(NC4);   // agg scratch 1
  float* t2f = (float*)carve(NC4);   // agg scratch 2

  _Float16* x16    = (_Float16*)carve(NC2);
  _Float16* t1x16  = (_Float16*)carve(NC2);
  _Float16* t2x16  = (_Float16*)carve(NC2);
  _Float16* h16    = (_Float16*)carve(NC2);
  _Float16* t1h16  = (_Float16*)carve(NC2);
  _Float16* t2h16  = (_Float16*)carve(NC2);
  _Float16* hr16   = (_Float16*)carve(NC2);
  _Float16* t1hr16 = (_Float16*)carve(NC2);
  _Float16* t2hr16 = (_Float16*)carve(NC2);

  // packed weights: 6 names x K=3, each 4*8*32*16 halves
  const size_t PKSZ = 4 * 8 * 512;               // halves per 128x128 matrix
  _Float16* pk[6][3];
  for (int nmi = 0; nmi < 6; ++nmi)
    for (int t = 0; t < 3; ++t)
      pk[nmi][t] = (_Float16*)carve(PKSZ * sizeof(_Float16));
  _Float16* pkp = (_Float16*)carve((size_t)4 * 4 * 512 * sizeof(_Float16));

  // -------- pack weights (loop-invariant; done per call, deterministic) ----
  for (int nmi = 0; nmi < 6; ++nmi)
    for (int t = 0; t < 3; ++t)
      pack_weight_kernel<<<(4 * 8 * 512 + 255) / 256, 256, 0, stream>>>(
          Wsrc[nmi] + (size_t)t * HID * HID, pk[nmi][t], 8);
  pack_weight_kernel<<<(4 * 4 * 512 + 255) / 256, 256, 0, stream>>>(Wp, pkp, 4);

  // -------- init h = 0 --------
  hipMemsetAsync(h, 0, NC4, stream);

  const int eGrid  = N_EDGES / 256;       // 1250
  const int aGrid  = N_EDGES / 8;         // 40000 (8 edges/block, 1 wave/edge)
  const int cGrid  = (int)(NC / 256);     // 5000
  const int gGrid  = N_NODES / 16;        // 625

  for (int t = 0; t < T_STEPS; ++t) {
    const float*     x   = feats + (size_t)t * NC;
    const long long* src = edges + (size_t)t * 2 * N_EDGES;
    const long long* dst = src + N_EDGES;

    // edge weights
    hipMemsetAsync(deg, 0, N_NODES * sizeof(float), stream);
    deg_kernel<<<eGrid, 256, 0, stream>>>(src, dst, deg);
    edgew_kernel<<<eGrid, 256, 0, stream>>>(src, dst, deg, ew);

    // x chebyshev basis
    cvt16_kernel<<<cGrid, 256, 0, stream>>>(x, x16, (int)NC);
    hipMemsetAsync(t1f, 0, NC4, stream);
    agg_kernel<<<aGrid, 256, 0, stream>>>(ew, src, dst, x, t1f);
    hipMemsetAsync(t2f, 0, NC4, stream);
    agg_kernel<<<aGrid, 256, 0, stream>>>(ew, src, dst, t1f, t2f);
    cvt16_kernel<<<cGrid, 256, 0, stream>>>(t1f, t1x16, (int)NC);
    cheb2_kernel<<<cGrid, 256, 0, stream>>>(t2f, x, t2x16, (int)NC);

    // h chebyshev basis
    cvt16_kernel<<<cGrid, 256, 0, stream>>>(h, h16, (int)NC);
    hipMemsetAsync(t1f, 0, NC4, stream);
    agg_kernel<<<aGrid, 256, 0, stream>>>(ew, src, dst, h, t1f);
    hipMemsetAsync(t2f, 0, NC4, stream);
    agg_kernel<<<aGrid, 256, 0, stream>>>(ew, src, dst, t1f, t2f);
    cvt16_kernel<<<cGrid, 256, 0, stream>>>(t1f, t1h16, (int)NC);
    cheb2_kernel<<<cGrid, 256, 0, stream>>>(t2f, h, t2h16, (int)NC);

    // z = sigmoid(cheb(x;Wxz) + cheb(h;Whz))
    gemm_fused6_kernel<<<gGrid, 256, 0, stream>>>(
        x16, t1x16, t2x16, h16, t1h16, t2h16,
        pk[0][0], pk[0][1], pk[0][2], pk[1][0], pk[1][1], pk[1][2],
        Bsrc[0], Bsrc[1], z, z, z, 0);

    // hr = h * sigmoid(cheb(x;Wxr) + cheb(h;Whr))
    gemm_fused6_kernel<<<gGrid, 256, 0, stream>>>(
        x16, t1x16, t2x16, h16, t1h16, t2h16,
        pk[2][0], pk[2][1], pk[2][2], pk[3][0], pk[3][1], pk[3][2],
        Bsrc[2], Bsrc[3], h, h, hr, 1);

    // hr chebyshev basis
    cvt16_kernel<<<cGrid, 256, 0, stream>>>(hr, hr16, (int)NC);
    hipMemsetAsync(t1f, 0, NC4, stream);
    agg_kernel<<<aGrid, 256, 0, stream>>>(ew, src, dst, hr, t1f);
    hipMemsetAsync(t2f, 0, NC4, stream);
    agg_kernel<<<aGrid, 256, 0, stream>>>(ew, src, dst, t1f, t2f);
    cvt16_kernel<<<cGrid, 256, 0, stream>>>(t1f, t1hr16, (int)NC);
    cheb2_kernel<<<cGrid, 256, 0, stream>>>(t2f, hr, t2hr16, (int)NC);

    // h = z*h + (1-z)*tanh(cheb(x;Wxh) + cheb(hr;Whh))  (in place)
    gemm_fused6_kernel<<<gGrid, 256, 0, stream>>>(
        x16, t1x16, t2x16, hr16, t1hr16, t2hr16,
        pk[4][0], pk[4][1], pk[4][2], pk[5][0], pk[5][1], pk[5][2],
        Bsrc[4], Bsrc[5], z, h, h, 2);
  }

  // out = h @ Wp + bp
  cvt16_kernel<<<cGrid, 256, 0, stream>>>(h, h16, (int)NC);
  gemm_proj_kernel<<<gGrid, 128, 0, stream>>>(h16, pkp, bp, out);
}
